// SteerablePocketBlock_87333864997035
// MI455X (gfx1250) — matrix operand order repaired and checked by reading, not compiled, not run on hardware
//
#include <hip/hip_runtime.h>

typedef __bf16 bf16_t;
typedef __attribute__((ext_vector_type(16))) __bf16 v16bf;
typedef __attribute__((ext_vector_type(8)))  float  v8f;
typedef int v4i_vs __attribute__((vector_size(4 * sizeof(int))));

constexpr int BB = 2;
constexpr int NN = 512;
constexpr int DD = 128;

#if __has_builtin(__builtin_amdgcn_global_load_async_to_lds_b128) && \
    __has_builtin(__builtin_amdgcn_s_wait_asynccnt)
#define HAS_ASYNC 1
#else
#define HAS_ASYNC 0
#endif

__device__ __forceinline__ float silu_f(float x) { return x / (1.0f + __expf(-x)); }
__device__ __forceinline__ float sigm_f(float x) { return 1.0f / (1.0f + __expf(-x)); }

#if HAS_ASYNC
__device__ __forceinline__ void async_cp16(const float* g, float* l) {
    __builtin_amdgcn_global_load_async_to_lds_b128(
        (__attribute__((address_space(1))) v4i_vs*)const_cast<float*>(g),
        (__attribute__((address_space(3))) v4i_vs*)l, 0, 0);
}
#endif

// ---------------------------------------------------------------------------
// Kernel 1: per-node separable pieces of edge-MLP layer 1.
//   base[b,n,d] = feat@eW1[0:128,d]   + cons*eW1[257,d] + eb1[d]
//   P   [b,n,d] = feat@eW1[128:256,d] + cons*eW1[258,d]
// ---------------------------------------------------------------------------
__global__ void precompute_kernel(const float* __restrict__ feat,
                                  const float* __restrict__ cons,
                                  const float* __restrict__ eW1,
                                  const float* __restrict__ eb1,
                                  float* __restrict__ wsP,
                                  float* __restrict__ wsBase) {
    int n = blockIdx.x, b = blockIdx.y, d = threadIdx.x;
    int row = b * NN + n;
    const float* f = feat + (size_t)row * DD;
    float cv = cons[row];
    float bacc = eb1[d] + cv * eW1[257 * DD + d];
    float pacc = cv * eW1[258 * DD + d];
    for (int k = 0; k < DD; ++k) {
        float fv = f[k];
        bacc += fv * eW1[k * DD + d];
        pacc += fv * eW1[(DD + k) * DD + d];
    }
    wsBase[(size_t)row * DD + d] = bacc;
    wsP[(size_t)row * DD + d] = pacc;
}

// ---------------------------------------------------------------------------
// Kernel 2: eW2 / vW1 -> transposed bf16 (W^T[n][k]) for contiguous B-frags.
// ---------------------------------------------------------------------------
__global__ void convert_kernel(const float* __restrict__ eW2,
                               const float* __restrict__ vW1,
                               bf16_t* __restrict__ wsEW2T,
                               bf16_t* __restrict__ wsVW1T) {
    for (int idx = threadIdx.x; idx < DD * DD; idx += blockDim.x) {
        int n = idx / DD, k = idx % DD;
        wsEW2T[idx] = (bf16_t)eW2[k * DD + n];
        wsVW1T[idx] = (bf16_t)vW1[k * DD + n];
    }
}

// ---------------------------------------------------------------------------
// Kernel 3: fused edge kernel. One block per (b, i); 8 wave32.
// ---------------------------------------------------------------------------
__global__ __launch_bounds__(256)
void edge_kernel(const float* __restrict__ feat, const float* __restrict__ vecf,
                 const float* __restrict__ pos,  const float* __restrict__ pmask,
                 const float* __restrict__ eW1,  const float* __restrict__ eb2,
                 const float* __restrict__ sW1g, const float* __restrict__ sb1g,
                 const float* __restrict__ sW2g, const float* __restrict__ sb2g,
                 const float* __restrict__ vb1g, const float* __restrict__ vW2g,
                 const float* __restrict__ vb2g,
                 const float* __restrict__ lnw,  const float* __restrict__ lnb,
                 const float* __restrict__ wsP,  const float* __restrict__ wsBase,
                 const bf16_t* __restrict__ wsEW2T, const bf16_t* __restrict__ wsVW1T,
                 float* __restrict__ outS, float* __restrict__ outV,
                 float* __restrict__ outE) {
    __shared__ bf16_t ldsEW2[DD * DD];   // eW2^T bf16 (32 KB)
    __shared__ bf16_t ldsVW1[DD * DD];   // vW1^T bf16 (32 KB)
    __shared__ float  sP[2][16][DD];     // double-buffered P tiles (16 KB)
    __shared__ bf16_t sA[16][DD];        // silu(h1) bf16  (A of GEMM1)
    __shared__ bf16_t sA2[16][DD];       // edge_hidden bf16 (A of GEMM2)
    __shared__ float  sEH[16][DD];       // edge_hidden f32 / gate partials
    __shared__ float  sBase[DD], sW256[DD], sW259[DD], sEb2[DD], sVb1[DD], sV2[DD];
    __shared__ float  sFeat[DD];
    __shared__ float  sDist[16], sMask[16], sEw[16];
    __shared__ float  sDir[16][3];
    __shared__ float  sPart[16][8];
    __shared__ float  sMsg[DD], sH[DD], sX[DD];
    __shared__ float  sVm[16][3];
    __shared__ float  sRed[2];

    const int i = blockIdx.x, b = blockIdx.y;
    const int t = threadIdx.x;
    const int lane = t & 31;
    const int wv = t >> 5;           // wave id 0..7
    const int hlf = lane >> 4;       // lane half
    const int r = lane & 15;
    const int ncol0 = wv * 16;
    const int rowI = b * NN + i;

    // Stage weights into LDS (32-bit copies).
    {
        const unsigned int* s1 = (const unsigned int*)wsEW2T;
        const unsigned int* s2 = (const unsigned int*)wsVW1T;
        unsigned int* d1 = (unsigned int*)ldsEW2;
        unsigned int* d2 = (unsigned int*)ldsVW1;
        for (int idx = t; idx < DD * DD / 2; idx += 256) { d1[idx] = s1[idx]; d2[idx] = s2[idx]; }
    }
    if (t < DD) {
        sBase[t] = wsBase[(size_t)rowI * DD + t];
        sW256[t] = eW1[256 * DD + t];
        sW259[t] = eW1[259 * DD + t];
        sEb2[t]  = eb2[t];
        sVb1[t]  = vb1g[t];
        sV2[t]   = vW2g[t];
        sFeat[t] = feat[(size_t)rowI * DD + t];
    }
    const float px = pos[rowI * 3 + 0], py = pos[rowI * 3 + 1], pz = pos[rowI * 3 + 2];
    float msg = 0.0f;                         // valid for t < 128
    float vmx = 0.0f, vmy = 0.0f, vmz = 0.0f; // valid for t < 16

#if HAS_ASYNC
    // Preload chunk 0 P-tile into buffer 0 (2 x b128 per thread).
    {
        const float* gbase = wsP + (size_t)(b * NN) * DD;
        async_cp16(gbase + (size_t)t * 4,         &sP[0][0][0] + (size_t)t * 4);
        async_cp16(gbase + (size_t)(t + 256) * 4, &sP[0][0][0] + (size_t)(t + 256) * 4);
    }
#endif
    __syncthreads();

    for (int j0 = 0; j0 < NN; j0 += 16) {
        const int buf = (j0 >> 4) & 1;
        // --- geometry for this 16-j tile ---
        if (t < 16) {
            int j = j0 + t;
            int rowJ = b * NN + j;
            float dx = px - pos[rowJ * 3 + 0];
            float dy = py - pos[rowJ * 3 + 1];
            float dz = pz - pos[rowJ * 3 + 2];
            float dd = fmaxf(sqrtf(dx * dx + dy * dy + dz * dz), 1e-8f);
            sDist[t] = dd;
            float inv = 1.0f / dd;
            sDir[t][0] = dx * inv; sDir[t][1] = dy * inv; sDir[t][2] = dz * inv;
            sMask[t] = pmask[(size_t)rowI * NN + j];
        }
#if HAS_ASYNC
        // Prefetch next chunk's P tile into the other buffer, then wait for ours.
        if (j0 + 16 < NN) {
            const float* gbase = wsP + (size_t)(b * NN + j0 + 16) * DD;
            float* lbase = &sP[buf ^ 1][0][0];
            async_cp16(gbase + (size_t)t * 4,         lbase + (size_t)t * 4);
            async_cp16(gbase + (size_t)(t + 256) * 4, lbase + (size_t)(t + 256) * 4);
            __builtin_amdgcn_s_wait_asynccnt(2);
        } else {
            __builtin_amdgcn_s_wait_asynccnt(0);
        }
#endif
        __syncthreads();
        // --- layer-1 (separated): h1 = silu(base_i + P_j + dist*w256 + mask*w259) ---
        #pragma unroll
        for (int q = 0; q < 8; ++q) {
            int idx = t + 256 * q;
            int jj = idx >> 7, d = idx & (DD - 1);
#if HAS_ASYNC
            float p = sP[buf][jj][d];
#else
            float p = wsP[(size_t)(b * NN + j0 + jj) * DD + d];
#endif
            float h = sBase[d] + p + sDist[jj] * sW256[d] + sMask[jj] * sW259[d];
            sA[jj][d] = (bf16_t)silu_f(h);
        }
        __syncthreads();
        // --- GEMM1: edge_hidden = sA @ eW2 + eb2 (bf16 WMMA, f32 acc) ---
        v8f c = {0.f, 0.f, 0.f, 0.f, 0.f, 0.f, 0.f, 0.f};
        for (int kb = 0; kb < DD; kb += 32) {
            union { v16bf v; unsigned int u[8]; } af, bg;
            #pragma unroll
            for (int v = 0; v < 8; ++v) {
                int ka = kb + 2 * (v & 3) + 8 * hlf + 16 * (v >> 2);
                af.u[v] = *(const unsigned int*)&sA[r][ka];
                int kbidx = kb + 2 * (v & 3) + 8 * hlf + 16 * (v >> 2);
                bg.u[v] = *(const unsigned int*)&ldsEW2[(ncol0 + r) * DD + kbidx];
            }
            c = __builtin_amdgcn_wmma_f32_16x16x32_bf16(false, af.v, false, bg.v,
                                                        (short)0, c, false, false);
        }
        #pragma unroll
        for (int v = 0; v < 8; ++v) {
            int row = v + 8 * hlf;
            int col = ncol0 + r;
            float eh = c[v] + sEb2[col];
            sEH[row][col] = eh;
            sA2[row][col] = (bf16_t)eh;
        }
        __syncthreads();
        // --- edge weights: sigmoid(mean_d edge_hidden) * mask (two-stage reduce) ---
        if (t < 128) {
            int row = t >> 3, seg = t & 7;
            float s = 0.0f;
            #pragma unroll
            for (int k = 0; k < 16; ++k) s += sEH[row][seg * 16 + k];
            sPart[row][seg] = s;
        }
        __syncthreads();
        if (t < 16) {
            float s = 0.0f;
            #pragma unroll
            for (int k = 0; k < 8; ++k) s += sPart[t][k];
            float ew = sigm_f(s * (1.0f / DD)) * sMask[t];
            sEw[t] = ew;
            outE[(size_t)rowI * NN + j0 + t] = ew;
        }
        __syncthreads();
        // --- scalar message accumulation (register-resident per d) ---
        if (t < DD) {
            float acc = msg;
            for (int jj = 0; jj < 16; ++jj) acc += sEw[jj] * sEH[jj][t];
            msg = acc;
        }
        __syncthreads();
        // --- GEMM2: gh = edge_hidden @ vW1 + vb1 ; partial = silu(gh)*vW2[col] ---
        v8f c2 = {0.f, 0.f, 0.f, 0.f, 0.f, 0.f, 0.f, 0.f};
        for (int kb = 0; kb < DD; kb += 32) {
            union { v16bf v; unsigned int u[8]; } af, bg;
            #pragma unroll
            for (int v = 0; v < 8; ++v) {
                int ka = kb + 2 * (v & 3) + 8 * hlf + 16 * (v >> 2);
                af.u[v] = *(const unsigned int*)&sA2[r][ka];
                int kbidx = kb + 2 * (v & 3) + 8 * hlf + 16 * (v >> 2);
                bg.u[v] = *(const unsigned int*)&ldsVW1[(ncol0 + r) * DD + kbidx];
            }
            c2 = __builtin_amdgcn_wmma_f32_16x16x32_bf16(false, af.v, false, bg.v,
                                                         (short)0, c2, false, false);
        }
        #pragma unroll
        for (int v = 0; v < 8; ++v) {
            int row = v + 8 * hlf;
            int col = ncol0 + r;
            float gh = c2[v] + sVb1[col];
            sEH[row][col] = silu_f(gh) * sV2[col];
        }
        __syncthreads();
        // --- vector gate reduce (two-stage) + vector message accumulation ---
        if (t < 128) {
            int row = t >> 3, seg = t & 7;
            float s = 0.0f;
            #pragma unroll
            for (int k = 0; k < 16; ++k) s += sEH[row][seg * 16 + k];
            sPart[row][seg] = s;
        }
        __syncthreads();
        if (t < 16) {
            float g = vb2g[0];
            #pragma unroll
            for (int k = 0; k < 8; ++k) g += sPart[t][k];
            float vg = g * sEw[t];
            vmx += vg * sDir[t][0];
            vmy += vg * sDir[t][1];
            vmz += vg * sDir[t][2];
        }
        __syncthreads();
    }

    // ---- tail: node update MLP + LayerNorm, vector output ----
    if (t < DD) sMsg[t] = msg;
    if (t < 16) { sVm[t][0] = vmx; sVm[t][1] = vmy; sVm[t][2] = vmz; }
    __syncthreads();
    if (t < DD) {
        float acc = sb1g[t];
        for (int k = 0; k < DD; ++k) acc += sFeat[k] * sW1g[k * DD + t];
        for (int k = 0; k < DD; ++k) acc += sMsg[k] * sW1g[(DD + k) * DD + t];
        sH[t] = silu_f(acc);
    }
    __syncthreads();
    if (t < DD) {
        float acc = sb2g[t];
        for (int k = 0; k < DD; ++k) acc += sH[k] * sW2g[k * DD + t];
        sX[t] = sFeat[t] + acc;
    }
    __syncthreads();
    if (t == 0) {
        float s = 0.0f, sq = 0.0f;
        for (int k = 0; k < DD; ++k) { float x = sX[k]; s += x; sq += x * x; }
        float mu = s * (1.0f / DD);
        float var = sq * (1.0f / DD) - mu * mu;
        sRed[0] = mu;
        sRed[1] = rsqrtf(var + 1e-5f);
        float v0 = 0.f, v1 = 0.f, v2 = 0.f;
        for (int k = 0; k < 16; ++k) { v0 += sVm[k][0]; v1 += sVm[k][1]; v2 += sVm[k][2]; }
        outV[rowI * 3 + 0] = vecf[rowI * 3 + 0] + v0;
        outV[rowI * 3 + 1] = vecf[rowI * 3 + 1] + v1;
        outV[rowI * 3 + 2] = vecf[rowI * 3 + 2] + v2;
    }
    __syncthreads();
    if (t < DD) {
        outS[(size_t)rowI * DD + t] = (sX[t] - sRed[0]) * sRed[1] * lnw[t] + lnb[t];
    }
}

extern "C" void kernel_launch(void* const* d_in, const int* in_sizes, int n_in,
                              void* d_out, int out_size, void* d_ws, size_t ws_size,
                              hipStream_t stream) {
    const float* feat  = (const float*)d_in[0];
    const float* vecf  = (const float*)d_in[1];
    const float* pos   = (const float*)d_in[2];
    const float* pmask = (const float*)d_in[3];
    const float* cons  = (const float*)d_in[4];
    const float* eW1   = (const float*)d_in[5];
    const float* eb1   = (const float*)d_in[6];
    const float* eW2   = (const float*)d_in[7];
    const float* eb2   = (const float*)d_in[8];
    const float* sW1   = (const float*)d_in[9];
    const float* sb1   = (const float*)d_in[10];
    const float* sW2   = (const float*)d_in[11];
    const float* sb2   = (const float*)d_in[12];
    const float* vW1   = (const float*)d_in[13];
    const float* vb1   = (const float*)d_in[14];
    const float* vW2   = (const float*)d_in[15];
    const float* vb2   = (const float*)d_in[16];
    const float* lnw   = (const float*)d_in[17];
    const float* lnb   = (const float*)d_in[18];

    // workspace layout
    float*  wsP    = (float*)d_ws;                 // B*N*D f32
    float*  wsBase = wsP + BB * NN * DD;           // B*N*D f32
    bf16_t* wsEW2T = (bf16_t*)(wsBase + BB * NN * DD);
    bf16_t* wsVW1T = wsEW2T + DD * DD;

    // output layout: scalar_out [B,N,D] | vector_out [B,N,3] | edge_weights [B,N,N]
    float* outS = (float*)d_out;
    float* outV = outS + BB * NN * DD;
    float* outE = outV + BB * NN * 3;

    dim3 gA(NN, BB);
    precompute_kernel<<<gA, DD, 0, stream>>>(feat, cons, eW1, eb1, wsP, wsBase);
    convert_kernel<<<1, 256, 0, stream>>>(eW2, vW1, wsEW2T, wsVW1T);
    dim3 gB(NN, BB);
    edge_kernel<<<gB, 256, 0, stream>>>(feat, vecf, pos, pmask, eW1, eb2,
                                        sW1, sb1, sW2, sb2, vb1, vW2, vb2,
                                        lnw, lnb, wsP, wsBase, wsEW2T, wsVW1T,
                                        outS, outV, outE);
}
